// FreeLoss_51805895524956
// MI455X (gfx1250) — compile-verified, matrix operand order checked by reading, not run on hardware
//
#include <hip/hip_runtime.h>
#include <hip/hip_bf16.h>

// ---------------- problem constants (from reference) ----------------
#define BI 16      // batch
#define PP 65536   // priors
#define NN 32      // truths per image
#define CC 81      // classes
#define KK 200     // TOPK
#define ROWS 128   // rows (priors) per k_neg block
#define SCALE_DOWN (1.0f/1024.0f)   // keep partials inside f16 range for WMMA reduce
#define SCALE_UP   1024.0f

typedef __attribute__((ext_vector_type(16))) _Float16 v16h;
typedef __attribute__((ext_vector_type(8)))  float    v8f;
typedef __attribute__((ext_vector_type(4)))  int      v4i;

// async-to-LDS path (gfx1250 ASYNCcnt): guarded, with portable fallback
#if defined(__has_builtin)
#  if __has_builtin(__builtin_amdgcn_global_load_async_to_lds_b128)
#    define HAS_ASYNC_LDS 1
#  endif
#  if __has_builtin(__builtin_amdgcn_s_wait_asynccnt)
#    define HAS_WAIT_ASYNC_BUILTIN 1
#  endif
#endif

#if defined(HAS_ASYNC_LDS)
typedef __attribute__((address_space(1))) v4i* v4i_gptr;   // global (AS1)
typedef __attribute__((address_space(3))) v4i* v4i_lptr;   // LDS (AS3)
#endif

// ---------------- workspace layout (bytes) ----------------
// [0,2048)       : t2 max bits, uint32 [BI*NN=512]
// [2048,8192)    : class masks, uint32 [BI*CC=1296]
// [8192,40960)   : neg partial sums, float [8192]   (multiple of 256)
// [40960,43008)  : pos partial sums, float [512]    (multiple of 256)
#define WS_T2    0
#define WS_MASK  2048
#define WS_NEGP  8192
#define WS_POSP  40960

__device__ __forceinline__ float iou_corners(float ax1, float ay1, float ax2, float ay2,
                                             float bx1, float by1, float bx2, float by2) {
    float lx = fmaxf(ax1, bx1), ly = fmaxf(ay1, by1);
    float rx = fminf(ax2, bx2), ry = fminf(ay2, by2);
    float w = fmaxf(rx - lx, 0.0f), h = fmaxf(ry - ly, 0.0f);
    float inter = w * h;
    float aa = (ax2 - ax1) * (ay2 - ay1);
    float ab = (bx2 - bx1) * (by2 - by1);
    return inter / (aa + ab - inter);
}

__device__ __forceinline__ float sl1_comp(float val) {
    float a = fabsf(val);
    return (a < 0.11f) ? (0.5f / 0.11f) * val * val : (a - 0.5f * 0.11f);
}

// ---------------- init: zero t2 accumulators, build per-(image,class) truth bitmasks ----
__global__ void k_init(const int* __restrict__ labels,
                       unsigned* __restrict__ t2bits, unsigned* __restrict__ mask) {
    int tid = threadIdx.x;
    for (int i = tid; i < BI * NN; i += 256) t2bits[i] = 0u;
    for (int i = tid; i < BI * CC; i += 256) {
        int b = i / CC, c = i % CC;
        unsigned m = 0u;
        for (int n = 0; n < NN; ++n)
            if (labels[b * NN + n] == c) m |= (1u << n);
        mask[i] = m;
    }
}

// ---------------- pass 1: t2[b,n] = max_p IoU(decoded(b,p), truth(b,n)) ----------------
__global__ __launch_bounds__(256) void k_t2(const float4* __restrict__ loc4,
                                            const float4* __restrict__ pri4,
                                            const float4* __restrict__ tr4,
                                            unsigned* __restrict__ t2bits) {
    __shared__ float    s_tr[NN][4];
    __shared__ unsigned s_max[NN];
    int tid = threadIdx.x;
    int b = blockIdx.x >> 6;          // 64 blocks per image
    int chunk = blockIdx.x & 63;
    if (tid < NN) {
        float4 t = tr4[b * NN + tid];
        s_tr[tid][0] = t.x; s_tr[tid][1] = t.y; s_tr[tid][2] = t.z; s_tr[tid][3] = t.w;
        s_max[tid] = 0u;
    }
    __syncthreads();
    float mx[NN];
#pragma unroll
    for (int n = 0; n < NN; ++n) mx[n] = 0.0f;
    for (int it = 0; it < 4; ++it) {
        int p = chunk * 1024 + it * 256 + tid;
        float4 l = loc4[(size_t)b * PP + p];
        float4 q = pri4[p];
        float cx = q.x + l.x * 0.1f * q.z;
        float cy = q.y + l.y * 0.1f * q.w;
        float w = q.z * __expf(l.z * 0.2f);
        float h = q.w * __expf(l.w * 0.2f);
        float ax1 = cx - 0.5f * w, ay1 = cy - 0.5f * h;
        float ax2 = cx + 0.5f * w, ay2 = cy + 0.5f * h;
#pragma unroll
        for (int n = 0; n < NN; ++n) {
            float iou = iou_corners(ax1, ay1, ax2, ay2,
                                    s_tr[n][0], s_tr[n][1], s_tr[n][2], s_tr[n][3]);
            mx[n] = fmaxf(mx[n], iou);
        }
    }
#pragma unroll
    for (int n = 0; n < NN; ++n) atomicMax(&s_max[n], __float_as_uint(mx[n]));
    __syncthreads();
    if (tid < NN) atomicMax(&t2bits[b * NN + tid], s_max[tid]);
}

// ---------------- pass 2 (HBM-dominant): conf staged global->LDS once, softmax + neg loss
// Tile = 128 rows x 81 classes x f32 = 41472 B: both start offset and length are 16B
// multiples, so staging uses aligned coalesced b128 (async-to-LDS when available).
__global__ __launch_bounds__(ROWS) void k_neg(const float4* __restrict__ loc4,
                                              const float* __restrict__ conf,
                                              const float4* __restrict__ pri4,
                                              const float4* __restrict__ tr4,
                                              const unsigned* __restrict__ t2bits,
                                              const unsigned* __restrict__ mask,
                                              float* __restrict__ negp) {
    __shared__ __align__(16) float s_conf[ROWS * CC];   // 41472 B
    __shared__ float    s_tr[NN][4];
    __shared__ float    s_den[NN];
    __shared__ unsigned s_mask[CC];
    __shared__ float    s_ioun[ROWS * 33];              // 33-stride: conflict-free dyn index
    __shared__ float    s_red[ROWS];
    int tid = threadIdx.x;
    int b = blockIdx.x >> 9;                 // 512 blocks per image
    int chunk = blockIdx.x & 511;
    int p = chunk * ROWS + tid;

    // ---- stage the conf tile into LDS (single global read of conf overall) ----
    size_t tile_base = ((size_t)b * PP + (size_t)chunk * ROWS) * (size_t)CC; // 16B aligned
    const float4* gsrc = (const float4*)(conf + tile_base);
    float4* lds4 = (float4*)s_conf;
    __builtin_prefetch(&conf[tile_base], 0, 1);          // gfx1250 global_prefetch_b8
    const int NV4 = ROWS * CC / 4;                       // 2592
    for (int i = tid; i < NV4; i += ROWS) {
#if defined(HAS_ASYNC_LDS)
        // builtin signature: (v4i AS1* src, v4i AS3* dst, imm offset, imm cpol)
        const v4i* gg_c = (const v4i*)(gsrc + i);
        v4i* gg = const_cast<v4i*>(gg_c);
        v4i* ll = (v4i*)(lds4 + i);
        __builtin_amdgcn_global_load_async_to_lds_b128(
            (v4i_gptr)gg, (v4i_lptr)ll, 0, 0);
#else
        lds4[i] = gsrc[i];
#endif
    }
#if defined(HAS_ASYNC_LDS)
#  if defined(HAS_WAIT_ASYNC_BUILTIN)
    __builtin_amdgcn_s_wait_asynccnt(0);
#  else
    asm volatile("s_wait_asynccnt 0x0" ::: "memory");
#  endif
#endif

    if (tid < NN) {
        float4 t = tr4[b * NN + tid];
        s_tr[tid][0] = t.x; s_tr[tid][1] = t.y; s_tr[tid][2] = t.z; s_tr[tid][3] = t.w;
        float t2 = fmaxf(__uint_as_float(t2bits[b * NN + tid]), 0.5f + 1e-12f);
        s_den[tid] = t2 - 0.5f;
    }
    if (tid < CC) s_mask[tid] = mask[b * CC + tid];
    __syncthreads();

    // decode (b,p) box once
    float4 l = loc4[(size_t)b * PP + p];
    float4 q = pri4[p];
    float cx = q.x + l.x * 0.1f * q.z;
    float cy = q.y + l.y * 0.1f * q.w;
    float w = q.z * __expf(l.z * 0.2f);
    float h = q.w * __expf(l.w * 0.2f);
    float ax1 = cx - 0.5f * w, ay1 = cy - 0.5f * h;
    float ax2 = cx + 0.5f * w, ay2 = cy + 0.5f * h;
#pragma unroll
    for (int n = 0; n < NN; ++n) {
        float iou = iou_corners(ax1, ay1, ax2, ay2,
                                s_tr[n][0], s_tr[n][1], s_tr[n][2], s_tr[n][3]);
        float v = (iou - 0.5f) / s_den[n];
        s_ioun[tid * 33 + n] = fminf(fmaxf(v, 0.0f), 1.0f);
    }

    // softmax over this thread's row in LDS (stride 81 is odd -> bank-conflict-free)
    const float* row = s_conf + tid * CC;
    float m = -3.402823466e38f;
    for (int c = 0; c < CC; ++c) m = fmaxf(m, row[c]);
    float s = 0.0f;
    for (int c = 0; c < CC; ++c) s += __expf(row[c] - m);
    float invs = 1.0f / s;
    // per-class prob, box_prob via truth-bitmask max, loss term
    float acc = 0.0f;
    for (int c = 0; c < CC; ++c) {
        float prob = __expf(row[c] - m) * invs;
        float bp = 0.0f;
        unsigned mk = s_mask[c];
        while (mk) {
            int n = __ffs(mk) - 1;
            bp = fmaxf(bp, s_ioun[tid * 33 + n]);
            mk &= mk - 1u;
        }
        float neg = prob * (1.0f - bp);
        acc += neg * neg * (-__logf(fmaxf(1.0f - neg, 1e-12f)));
    }
    s_red[tid] = acc;
    __syncthreads();
    for (int off = ROWS / 2; off > 0; off >>= 1) {
        if (tid < off) s_red[tid] += s_red[tid + off];
        __syncthreads();
    }
    if (tid == 0) negp[blockIdx.x] = s_red[0] * SCALE_DOWN;
}

// ---------------- pass 3: per-(b,n) exact top-200 (radix select) + positive bag loss ---
__global__ __launch_bounds__(256) void k_pos(const float4* __restrict__ loc4,
                                             const float* __restrict__ conf,
                                             const float4* __restrict__ pri4,
                                             const float4* __restrict__ tr4,
                                             const int* __restrict__ labels,
                                             float* __restrict__ posp) {
    __shared__ float    s_tr[NN][4];
    __shared__ unsigned s_hist[256];
    __shared__ unsigned long long s_key[256];
    __shared__ float s_r1[256], s_r2[256];
    __shared__ unsigned long long s_thr;
    __shared__ unsigned s_rem;
    __shared__ int s_cnt;

    int tid = threadIdx.x;
    int b = blockIdx.x >> 5;    // 32 (b,n) blocks per image
    int n = blockIdx.x & 31;
    if (tid < NN) {
        float4 t = tr4[b * NN + tid];
        s_tr[tid][0] = t.x; s_tr[tid][1] = t.y; s_tr[tid][2] = t.z; s_tr[tid][3] = t.w;
    }
    if (tid == 0) { s_thr = 0ull; s_rem = KK; s_cnt = 0; }
    __syncthreads();
    float tx1 = s_tr[n][0], ty1 = s_tr[n][1], tx2 = s_tr[n][2], ty2 = s_tr[n][3];

    // composite 48-bit key: (iou_bits << 16) | (0xFFFF - p)  -> unique per p,
    // descending order == value desc then index asc (matches lax.top_k ties).
    // 6 radix passes of 8 bits: fully deterministic, tie-free selection.
    for (int pass = 0; pass < 6; ++pass) {
        int shift = 40 - 8 * pass;
        s_hist[tid] = 0u;
        __syncthreads();
        unsigned long long thr = s_thr;
        for (int p = tid; p < PP; p += 256) {
            float4 q = pri4[p];
            float px1 = q.x - 0.5f * q.z, py1 = q.y - 0.5f * q.w;
            float px2 = q.x + 0.5f * q.z, py2 = q.y + 0.5f * q.w;
            float v = iou_corners(px1, py1, px2, py2, tx1, ty1, tx2, ty2);
            unsigned long long comp =
                ((unsigned long long)__float_as_uint(v) << 16) | (unsigned)(0xFFFFu - p);
            if ((comp >> (shift + 8)) == (thr >> (shift + 8)))
                atomicAdd(&s_hist[(unsigned)(comp >> shift) & 0xFFu], 1u);
        }
        __syncthreads();
        if (tid == 0) {
            unsigned rem = s_rem, acc = 0;
            int bin = 255;
            for (; bin >= 0; --bin) {
                unsigned hc = s_hist[bin];
                if (acc + hc >= rem) break;
                acc += hc;
            }
            s_thr = s_thr | ((unsigned long long)(unsigned)bin << shift);
            s_rem = s_rem - acc;
        }
        __syncthreads();
    }
    // collect: keys are unique -> exactly KK elements have comp >= thr
    unsigned long long thr = s_thr;
    s_key[tid] = 0ull;
    __syncthreads();
    for (int p = tid; p < PP; p += 256) {
        float4 q = pri4[p];
        float px1 = q.x - 0.5f * q.z, py1 = q.y - 0.5f * q.w;
        float px2 = q.x + 0.5f * q.z, py2 = q.y + 0.5f * q.w;
        float v = iou_corners(px1, py1, px2, py2, tx1, ty1, tx2, ty2);
        unsigned long long comp =
            ((unsigned long long)__float_as_uint(v) << 16) | (unsigned)(0xFFFFu - p);
        if (comp >= thr) {
            int pos = atomicAdd(&s_cnt, 1);
            if (pos < 256) s_key[pos] = comp;
        }
    }
    __syncthreads();
    // bitonic sort 256 keys descending (rank k == lax.top_k rank k)
    for (unsigned kk = 2; kk <= 256; kk <<= 1) {
        for (unsigned j = kk >> 1; j > 0; j >>= 1) {
            unsigned ixj = (unsigned)tid ^ j;
            if (ixj > (unsigned)tid) {
                bool desc = ((tid & kk) == 0);
                unsigned long long a = s_key[tid], c = s_key[ixj];
                if (desc ? (a < c) : (a > c)) { s_key[tid] = c; s_key[ixj] = a; }
            }
            __syncthreads();
        }
    }
    // bag loss over sorted top-200
    int lb = labels[b * NN + n];
    float wv = 0.0f, wx = 0.0f;
    if (tid < KK) {
        unsigned p = 0xFFFFu - (unsigned)(s_key[tid] & 0xFFFFull);
        size_t base = ((size_t)b * PP + p) * (size_t)CC;
        float m = -3.402823466e38f;
        for (int c = 0; c < CC; ++c) m = fmaxf(m, conf[base + c]);
        float ss = 0.0f;
        for (int c = 0; c < CC; ++c) ss += __expf(conf[base + c] - m);
        float mcls = __expf(conf[base + lb] - m) / ss;
        // reference's tile() pairing: truth index = (n*TOPK + k) % N
        int j = (n * KK + tid) & (NN - 1);
        float jx1 = s_tr[j][0], jy1 = s_tr[j][1], jx2 = s_tr[j][2], jy2 = s_tr[j][3];
        float4 q = pri4[p];
        float gcx = ((jx1 + jx2) * 0.5f - q.x) / (0.1f * q.z);
        float gcy = ((jy1 + jy2) * 0.5f - q.y) / (0.1f * q.w);
        float gw = __logf(fmaxf((jx2 - jx1) / q.z, 1e-12f)) * 5.0f;
        float gh = __logf(fmaxf((jy2 - jy1) / q.w, 1e-12f)) * 5.0f;
        float4 l = loc4[(size_t)b * PP + p];
        float reg = 0.75f * (sl1_comp(gcx - l.x) + sl1_comp(gcy - l.y) +
                             sl1_comp(gw - l.z) + sl1_comp(gh - l.w));
        float x = mcls * __expf(-reg);
        float ww = 1.0f / fmaxf(1.0f - x, 1e-12f);
        wv = ww; wx = ww * x;
    }
    s_r1[tid] = wv; s_r2[tid] = wx;
    __syncthreads();
    for (int off = 128; off > 0; off >>= 1) {
        if (tid < off) { s_r1[tid] += s_r1[tid + off]; s_r2[tid] += s_r2[tid + off]; }
        __syncthreads();
    }
    if (tid == 0) {
        float bag = s_r2[0] / s_r1[0];
        posp[blockIdx.x] = -__logf(fmaxf(bag, 1e-12f)) * SCALE_DOWN;
    }
}

// ---------------- final: deterministic WMMA tree reduction of partials ----------------
// D = A(16x32 f16) x ones(32x16) + C accumulates 512 halves = 256 f32 partials per
// v_wmma_f32_16x16x32_f16. f32 precision kept by hi/lo f16 split of each partial.
__device__ __forceinline__ float wmma_sum(const float* __restrict__ data, int count, int lane) {
    v16h ones;
#pragma unroll
    for (int i = 0; i < 16; ++i) ones[i] = (_Float16)1.0f;
    v8f acc = {0.f, 0.f, 0.f, 0.f, 0.f, 0.f, 0.f, 0.f};
    for (int base = 0; base < count; base += 256) {
        v16h a;
#pragma unroll
        for (int i = 0; i < 8; ++i) {
            float v = data[base + lane * 8 + i];
            _Float16 hi = (_Float16)v;
            _Float16 lo = (_Float16)(v - (float)hi);
            a[2 * i] = hi;
            a[2 * i + 1] = lo;
        }
        acc = __builtin_amdgcn_wmma_f32_16x16x32_f16(
            /*neg_a=*/false, a, /*neg_b=*/false, ones,
            /*c_mod=*/(short)0, acc, /*reuse_a=*/false, /*reuse_b=*/false);
    }
    // D[:, n]: lane l holds column n=l&15, rows M = vgpr(+8 for upper lanes).
    float s = 0.0f;
#pragma unroll
    for (int i = 0; i < 8; ++i) s += acc[i];
    return __shfl(s, 0, 32) + __shfl(s, 16, 32);   // rows 0-7 + rows 8-15 of column 0
}

__global__ __launch_bounds__(32) void k_finalize(const float* __restrict__ negp,
                                                 const float* __restrict__ posp,
                                                 float* __restrict__ out) {
    int lane = threadIdx.x;
    float totn = wmma_sum(negp, BI * 512, lane);   // 8192 partials
    float totp = wmma_sum(posp, 512, lane);
    if (lane == 0) {
        out[0] = totp * SCALE_UP / (float)(BI * NN) * 0.5f;                 // loss_p, ALPHA=0.5
        out[1] = totn * SCALE_UP / ((float)(BI * NN) * (float)KK) * 0.5f;   // loss_n, 1-ALPHA=0.5
    }
}

// ---------------- host launcher ----------------
extern "C" void kernel_launch(void* const* d_in, const int* in_sizes, int n_in,
                              void* d_out, int out_size, void* d_ws, size_t ws_size,
                              hipStream_t stream) {
    (void)in_sizes; (void)n_in; (void)out_size; (void)ws_size;
    const float4* loc4 = (const float4*)d_in[0];
    const float*  conf = (const float*)d_in[1];
    const float4* pri4 = (const float4*)d_in[2];
    const float4* tr4  = (const float4*)d_in[3];
    const int*    lab  = (const int*)d_in[4];
    char* ws = (char*)d_ws;
    unsigned* t2bits = (unsigned*)(ws + WS_T2);
    unsigned* mask   = (unsigned*)(ws + WS_MASK);
    float*    negp   = (float*)(ws + WS_NEGP);
    float*    posp   = (float*)(ws + WS_POSP);
    float*    out    = (float*)d_out;

    k_init<<<1, 256, 0, stream>>>(lab, t2bits, mask);
    k_t2<<<BI * 64, 256, 0, stream>>>(loc4, pri4, tr4, t2bits);
    k_neg<<<BI * 512, ROWS, 0, stream>>>(loc4, conf, pri4, tr4, t2bits, mask, negp);
    k_pos<<<BI * NN, 256, 0, stream>>>(loc4, conf, pri4, tr4, lab, posp);
    k_finalize<<<1, 32, 0, stream>>>(negp, posp, out);
}
